// ScaledDotProductAttention_7017976562065
// MI455X (gfx1250) — matrix operand order repaired and checked by reading, not compiled, not run on hardware
//
#include <hip/hip_runtime.h>

#define BATCH 4
#define SEQ   4096
#define DIM   512

typedef _Float16 h16;
typedef unsigned int u32;
typedef unsigned long long u64;
typedef __attribute__((ext_vector_type(16))) _Float16 v16h;
typedef __attribute__((ext_vector_type(8)))  _Float16 h8;
typedef __attribute__((ext_vector_type(4)))  _Float16 h4;
typedef __attribute__((ext_vector_type(8)))  float    v8f;
typedef __attribute__((ext_vector_type(4)))  float    f4;
typedef __attribute__((ext_vector_type(4)))  u32      u32x4;
typedef __attribute__((ext_vector_type(8)))  int      i32x8;
typedef __attribute__((ext_vector_type(4)))  int      i32x4;

#define WMMA_F16(a, b, c) \
  __builtin_amdgcn_wmma_f32_16x16x32_f16(false, (a), false, (b), (short)0, (c), false, false)

__device__ __forceinline__ v16h cat8(h8 lo, h8 hi) {
  return __builtin_shufflevector(lo, hi, 0,1,2,3,4,5,6,7,8,9,10,11,12,13,14,15);
}

__device__ __forceinline__ v8f vzero8() {
  v8f z = {0.f,0.f,0.f,0.f,0.f,0.f,0.f,0.f};
  return z;
}

// ---------------------------------------------------------------------------
// Tensor Data Mover: 2D f16 tile (tile_d1 rows x tile_d0 elems, row stride
// stride0 elems) from global memory into contiguous LDS at lds_off.
// D# built per CDNA5 ISA 8.3-8.6: group0 = {count=1, lds_addr, global_addr,
// type=2}; group1 = {data_size=2B, tensor dims, tile dims, dim0 stride}.
// Groups 2/3 (and the trailing group) zero: <=2D tensor.  Tracked on
// TENSORcnt.  This toolchain exposes the 6-arg builtin form.
// ---------------------------------------------------------------------------
__device__ __forceinline__ void tdm_load_2d(u32 lds_off, const void* gptr,
                                            u32 tile_d0, u32 tile_d1,
                                            u32 tensor_d0, u32 tensor_d1,
                                            u32 stride0) {
  const u64 ga = (u64)(size_t)gptr;
  u32x4 g0;
  g0[0] = 1u;                                   // count=1, user descriptor
  g0[1] = lds_off;                              // lds_addr (bytes)
  g0[2] = (u32)ga;                              // global_addr[31:0]
  g0[3] = (u32)(ga >> 32) | (2u << 30);         // global_addr[56:32] | type=2
  i32x8 g1;
  g1[0] = (int)(1u << 16);                      // data_size=1 (2 bytes)
  g1[1] = (int)((tensor_d0 & 0xFFFFu) << 16);   // tensor_dim0[15:0]
  g1[2] = (int)(((tensor_d0 >> 16) & 0xFFFFu) | ((tensor_d1 & 0xFFFFu) << 16));
  g1[3] = (int)(((tensor_d1 >> 16) & 0xFFFFu) | ((tile_d0 & 0xFFFFu) << 16));
  g1[4] = (int)(tile_d1 & 0xFFFFu);             // tile_dim1 (tile_dim2=0)
  g1[5] = (int)stride0;                         // tensor_dim0_stride[31:0]
  g1[6] = 0;                                    // stride hi / dim1_stride
  g1[7] = 0;
  i32x4 z4 = {0, 0, 0, 0};
  i32x8 z8 = {0, 0, 0, 0, 0, 0, 0, 0};
  __builtin_amdgcn_tensor_load_to_lds(g0, g1, z4, z4, z8, 0);
}

// ---------------------------------------------------------------------------
// Kernel 0: one-shot fp32 -> f16 conversion of the three weight matrices.
// ---------------------------------------------------------------------------
__global__ __launch_bounds__(256) void convert_w_kernel(
    const float* __restrict__ Wq, const float* __restrict__ Wk,
    const float* __restrict__ Wv, h16* __restrict__ wh) {
  const int p = blockIdx.y;
  const float* __restrict__ W = (p == 0) ? Wq : (p == 1) ? Wk : Wv;
  h16* __restrict__ o = wh + (size_t)p * DIM * DIM;
  const int i = blockIdx.x * blockDim.x + threadIdx.x;   // float4 index
  f4 x = ((const f4*)W)[i];
  h4 hv;
  hv[0] = (h16)x[0]; hv[1] = (h16)x[1]; hv[2] = (h16)x[2]; hv[3] = (h16)x[3];
  ((h4*)o)[i] = hv;
}

// ---------------------------------------------------------------------------
// Kernel 1: fused QKV projection, X @ W^T + b with f16 WMMA.
// blockIdx.x: 16-row tile of flattened [B*S, D]; blockIdx.y: 0=Q,1=K,2=V.
// Q/K stored row-major f16; V stored transposed [B, D, S] f16.
// ---------------------------------------------------------------------------
__global__ __launch_bounds__(128) void qkv_proj_kernel(
    const float* __restrict__ q_in, const float* __restrict__ k_in,
    const float* __restrict__ v_in, const h16* __restrict__ wh,
    const float* __restrict__ bq, const float* __restrict__ bk,
    const float* __restrict__ bv,
    h16* __restrict__ qh, h16* __restrict__ kh, h16* __restrict__ vT) {
  const int p = blockIdx.y;
  const float* __restrict__ X  = (p == 0) ? q_in : (p == 1) ? k_in : v_in;
  const h16*   __restrict__ Wp = wh + (size_t)p * DIM * DIM;
  const float* __restrict__ Bv = (p == 0) ? bq : (p == 1) ? bk : bv;

  const int row0 = blockIdx.x * 16;
  __shared__ _Float16 sX[16 * DIM];              // 16 KB activation tile

  const f4* __restrict__ X4 = (const f4*)(X + (size_t)row0 * DIM);
  for (int i = threadIdx.x; i < 16 * DIM / 4; i += blockDim.x) {
    f4 xv = X4[i];
    h4 hv;
    hv[0] = (h16)xv[0]; hv[1] = (h16)xv[1];
    hv[2] = (h16)xv[2]; hv[3] = (h16)xv[3];
    *(h4*)(sX + i * 4) = hv;
  }
  __syncthreads();

  const int lane = threadIdx.x & 31;
  const int wv   = threadIdx.x >> 5;
  const int n16  = lane & 15;
  const int hh   = lane >> 4;
  const int d0   = wv * 128;

  v8f acc[8];
#pragma unroll
  for (int nt = 0; nt < 8; ++nt) acc[nt] = vzero8();

#pragma unroll 4
  for (int kk = 0; kk < 16; ++kk) {
    const int k0 = kk * 32;
    h8 alo = *(const h8*)(sX + n16 * DIM + k0 + hh * 8);
    h8 ahi = *(const h8*)(sX + n16 * DIM + k0 + 16 + hh * 8);
    v16h a = cat8(alo, ahi);
#pragma unroll
    for (int nt = 0; nt < 8; ++nt) {
      const h16* wp = Wp + (size_t)(d0 + nt * 16 + n16) * DIM + k0 + hh * 8;
      v16h bf = cat8(*(const h8*)wp, *(const h8*)(wp + 16));
      acc[nt] = WMMA_F16(a, bf, acc[nt]);
    }
  }

  if (p < 2) {
    h16* __restrict__ o = (p == 0) ? qh : kh;
#pragma unroll
    for (int nt = 0; nt < 8; ++nt) {
      const float bias = Bv[d0 + nt * 16 + n16];
#pragma unroll
      for (int r = 0; r < 8; ++r) {
        const int m = r + 8 * hh;
        o[(size_t)(row0 + m) * DIM + d0 + nt * 16 + n16] = (h16)(acc[nt][r] + bias);
      }
    }
  } else {
    const int b  = row0 >> 12;
    const int s0 = row0 & (SEQ - 1);
#pragma unroll
    for (int nt = 0; nt < 8; ++nt) {
      const float bias = Bv[d0 + nt * 16 + n16];
      h8 st;
#pragma unroll
      for (int r = 0; r < 8; ++r) st[r] = (h16)(acc[nt][r] + bias);
      *(h8*)(vT + ((size_t)b * DIM + d0 + nt * 16 + n16) * SEQ + s0 + 8 * hh) = st;
    }
  }
}

// ---------------------------------------------------------------------------
// Kernel 2: causal flash attention.  2 waves/block own a 32-query tile.
// All Q/K/V operand traffic staged through LDS by the Tensor Data Mover;
// WMMA fragments are ds_load_b128 gathers from the staged tiles.
// ---------------------------------------------------------------------------
__global__ __launch_bounds__(64) void flash_attn_kernel(
    const h16* __restrict__ qh, const h16* __restrict__ kh,
    const h16* __restrict__ vT, const int* __restrict__ token_ids,
    float* __restrict__ out) {
  __shared__ _Float16 sQ[32 * DIM];     // 32 KB: 32 query rows
  __shared__ _Float16 sK[32 * DIM];     // 32 KB: 32-key chunk
  __shared__ _Float16 sV[DIM * 32];     // 32 KB: V^T slice [512 dims x 32 keys]
  __shared__ _Float16 sP[2][16 * 32];   //  2 KB: C->A relayout staging

  const int wv   = threadIdx.x >> 5;
  const int lane = threadIdx.x & 31;
  const int n16  = lane & 15;
  const int hh   = lane >> 4;
  const int b    = blockIdx.y;
  const int q0blk = blockIdx.x * 32;
  const int q0s   = q0blk + wv * 16;
  const size_t rowbase = (size_t)b * SEQ + q0s;

  const h16* __restrict__ khb = kh + (size_t)b * SEQ * DIM;
  const h16* __restrict__ vTb = vT + (size_t)b * DIM * SEQ;
  const int* __restrict__ tok = token_ids + (size_t)b * SEQ;
  const int nCh = ((q0blk + 31) >> 5) + 1;       // causal chunk bound (block)

  // TDM-stage the Q tile and the first K/V chunk (wave 0 issues).
  if (wv == 0) {
    tdm_load_2d((u32)(size_t)sQ, qh + ((size_t)b * SEQ + q0blk) * DIM,
                512, 32, 512, 32, 512);
    tdm_load_2d((u32)(size_t)sK, khb, 512, 32, 512, 32, 512);
    tdm_load_2d((u32)(size_t)sV, vTb, 32, 512, 32, 512, SEQ);
    __builtin_amdgcn_s_wait_tensorcnt(0);
  }
  __syncthreads();

  v8f acc[32];
#pragma unroll
  for (int nt = 0; nt < 32; ++nt) acc[nt] = vzero8();
  float Mr[8], Lr[8];
#pragma unroll
  for (int r = 0; r < 8; ++r) { Mr[r] = -1e30f; Lr[r] = 0.f; }

  const float scale = 0.044194173824159216f;     // 1/sqrt(512)
  const _Float16* sq = sQ + (wv * 16) * DIM;

  for (int j = 0; j < nCh; ++j) {
    const int kb = j * 32;
    const int col0 = kb + n16, col1 = col0 + 16;
    const bool pad0 = (tok[col0] == 0);
    const bool pad1 = (tok[col1] == 0);

    // scores S = Q K^T for this 16x32 key chunk (K=512 -> 16 WMMA steps).
    v8f c0 = vzero8(), c1 = vzero8();
#pragma unroll
    for (int kk = 0; kk < 16; ++kk) {
      const int k0 = kk * 32;
      h8 alo = *(const h8*)(sq + n16 * DIM + k0 + hh * 8);
      h8 ahi = *(const h8*)(sq + n16 * DIM + k0 + 16 + hh * 8);
      v16h a = cat8(alo, ahi);
      const _Float16* kp0 = sK + n16 * DIM + k0 + hh * 8;
      const _Float16* kp1 = sK + (n16 + 16) * DIM + k0 + hh * 8;
      v16h b0 = cat8(*(const h8*)kp0, *(const h8*)(kp0 + 16));
      v16h b1 = cat8(*(const h8*)kp1, *(const h8*)(kp1 + 16));
      c0 = WMMA_F16(a, b0, c0);
      c1 = WMMA_F16(a, b1, c1);
    }

    // mask + online softmax (row stats live in the 16-lane halves).
    float p0[8], p1[8], alpha[8];
#pragma unroll
    for (int r = 0; r < 8; ++r) {
      const int grow = q0s + r + 8 * hh;
      float s0 = c0[r] * scale; if (pad0 || col0 > grow) s0 = -1e30f;
      float s1 = c1[r] * scale; if (pad1 || col1 > grow) s1 = -1e30f;
      float t = fmaxf(s0, s1);
      t = fmaxf(t, __shfl_xor(t, 1, 32));
      t = fmaxf(t, __shfl_xor(t, 2, 32));
      t = fmaxf(t, __shfl_xor(t, 4, 32));
      t = fmaxf(t, __shfl_xor(t, 8, 32));
      const float newM = fmaxf(Mr[r], t);
      const float a_ = __expf(Mr[r] - newM);
      const float e0 = __expf(s0 - newM);
      const float e1 = __expf(s1 - newM);
      float rs = e0 + e1;
      rs += __shfl_xor(rs, 1, 32);
      rs += __shfl_xor(rs, 2, 32);
      rs += __shfl_xor(rs, 4, 32);
      rs += __shfl_xor(rs, 8, 32);
      Lr[r] = Lr[r] * a_ + rs;
      Mr[r] = newM;
      alpha[r] = a_;
      p0[r] = e0; p1[r] = e1;
    }
#pragma unroll
    for (int nt = 0; nt < 32; ++nt)
#pragma unroll
      for (int r = 0; r < 8; ++r) acc[nt][r] *= alpha[r];

    // Relayout P from C-layout to A-layout via LDS (same-wave DS ordering).
    _Float16* sp = sP[wv];
#pragma unroll
    for (int r = 0; r < 8; ++r) {
      const int m = r + 8 * hh;
      sp[m * 32 + n16]      = (h16)p0[r];
      sp[m * 32 + 16 + n16] = (h16)p1[r];
    }
    asm volatile("s_wait_dscnt 0x0" ::: "memory");
    h8 plo = *(const h8*)(sp + n16 * 32 + hh * 8);
    h8 phi = *(const h8*)(sp + n16 * 32 + 16 + hh * 8);
    v16h pf = cat8(plo, phi);

    // O += P V from the staged V^T slice.
#pragma unroll
    for (int nt = 0; nt < 32; ++nt) {
      const _Float16* vp = sV + (nt * 16 + n16) * 32 + hh * 8;
      v16h bf = cat8(*(const h8*)vp, *(const h8*)(vp + 16));
      acc[nt] = WMMA_F16(pf, bf, acc[nt]);
    }

    // Advance the TDM pipeline: refill K/V for the next chunk.
    __syncthreads();                    // everyone done reading sK/sV
    if (j + 1 < nCh) {
      if (wv == 0) {
        tdm_load_2d((u32)(size_t)sK, khb + (size_t)(kb + 32) * DIM,
                    512, 32, 512, 32, 512);
        tdm_load_2d((u32)(size_t)sV, vTb + (kb + 32), 32, 512, 32, 512, SEQ);
        __builtin_amdgcn_s_wait_tensorcnt(0);
      }
      __syncthreads();                  // data staged before next compute
    }
  }

  // Normalize and store fp32 output.
  float inv[8];
#pragma unroll
  for (int r = 0; r < 8; ++r) inv[r] = (Lr[r] > 0.f) ? 1.f / Lr[r] : 0.f;
  float* __restrict__ op = out + rowbase * DIM;
#pragma unroll
  for (int nt = 0; nt < 32; ++nt)
#pragma unroll
    for (int r = 0; r < 8; ++r)
      op[(size_t)(r + 8 * hh) * DIM + nt * 16 + n16] = acc[nt][r] * inv[r];
}

// ---------------------------------------------------------------------------
extern "C" void kernel_launch(void* const* d_in, const int* in_sizes, int n_in,
                              void* d_out, int out_size, void* d_ws, size_t ws_size,
                              hipStream_t stream) {
  const float* q_in = (const float*)d_in[0];
  const float* k_in = (const float*)d_in[1];
  const float* v_in = (const float*)d_in[2];
  const float* Wq   = (const float*)d_in[3];
  const float* bq   = (const float*)d_in[4];
  const float* Wk   = (const float*)d_in[5];
  const float* bk   = (const float*)d_in[6];
  const float* Wv   = (const float*)d_in[7];
  const float* bv   = (const float*)d_in[8];
  const int*   tok  = (const int*)d_in[9];

  // Workspace: q,k f16 [B*S,D]; v^T f16 [B,D,S]; weights f16 [3,D,D].
  const size_t NQ = (size_t)BATCH * SEQ * DIM;
  h16* qh = (h16*)d_ws;
  h16* kh = qh + NQ;
  h16* vT = kh + NQ;
  h16* wh = vT + NQ;

  dim3 g0(DIM * DIM / (4 * 256), 3);
  convert_w_kernel<<<g0, 256, 0, stream>>>(Wq, Wk, Wv, wh);

  dim3 g1(BATCH * SEQ / 16, 3);
  qkv_proj_kernel<<<g1, 128, 0, stream>>>(q_in, k_in, v_in, wh, bq, bk, bv,
                                          qh, kh, vT);

  dim3 g2(SEQ / 32, BATCH);
  flash_attn_kernel<<<g2, 64, 0, stream>>>(qh, kh, vT, tok, (float*)d_out);
}
